// AutoEncoder_5634997092513
// MI455X (gfx1250) — matrix operand order repaired and checked by reading, not compiled, not run on hardware
//
#include <hip/hip_runtime.h>
#include <hip/hip_bf16.h>

// ---------------------------------------------------------------------------
// CDNA5 (gfx1250) implementation: everything heavy runs through a bf16 WMMA
// GEMM (v_wmma_f32_16x16x32_bf16), wave32, 8 waves/block, 64x128x32 tiles.
// ---------------------------------------------------------------------------

typedef __bf16 bf16_t;
typedef __attribute__((ext_vector_type(16))) __bf16 v16bf;
typedef __attribute__((ext_vector_type(8)))  float  v8f;

#define BATCH 4
#define SEQ   2048
#define BM 64
#define BN 128
#define BK 32

__device__ __forceinline__ float geluf(float x) {
    return 0.5f * x * (1.0f + erff(x * 0.70710678118654752f));
}

static inline long cdivl(long a, long b) { return (a + b - 1) / b; }

// ---------------------------------------------------------------------------
// Generic batched bf16 GEMM:  C[b][m][n] = sum_k A[b][m][k]*B[b][k][n] (+bias[m])
// A element at a_rs*m + a_cs*k (+a_bs*b); B at b_rs*k + b_cs*n (+b_bs*b);
// C at c_rs*m + c_cs*n (+c_bs*b).  act: 0=none, 1=erf-GELU.
// If Cb != null, result is written bf16 to Cb, else fp32 to C.
// ---------------------------------------------------------------------------
__global__ __launch_bounds__(256)
void k_gemm_bf16(const bf16_t* __restrict__ A, const bf16_t* __restrict__ Bm,
                 float* __restrict__ C, bf16_t* __restrict__ Cb,
                 const float* __restrict__ bias,
                 int M, int Nc, int K,
                 long a_rs, long a_cs, long a_bs,
                 long b_rs, long b_cs, long b_bs,
                 long c_rs, long c_cs, long c_bs,
                 int act)
{
    __shared__ __attribute__((aligned(16))) bf16_t As[BM][BK];   // [m][k]
    __shared__ __attribute__((aligned(16))) bf16_t Bs[BN][BK];   // [n][k] (k-contiguous)

    const int tid  = threadIdx.x;
    const int wid  = tid >> 5;
    const int lane = tid & 31;
    const int bz   = blockIdx.z;
    const int m0   = blockIdx.y * BM;
    const int n0   = blockIdx.x * BN;

    A  += (long)bz * a_bs;
    Bm += (long)bz * b_bs;

    const int wm   = (wid >> 2) * 32;   // 0 or 32
    const int wn   = (wid & 3) * 32;    // 0,32,64,96
    const int half = lane >> 4;         // K half-wave select
    const int l16  = lane & 15;

    v8f acc[2][2] = {};

    for (int k0 = 0; k0 < K; k0 += BK) {
        // stage A tile (64x32) - 8 elements/thread, generic strides, zero-padded
        for (int i = tid; i < BM * BK; i += 256) {
            int m = i >> 5, k = i & 31;
            int gm = m0 + m, gk = k0 + k;
            As[m][k] = (gm < M && gk < K) ? A[(long)gm * a_rs + (long)gk * a_cs]
                                          : (bf16_t)0.0f;
        }
        // stage B tile transposed (128x32) - 16 elements/thread
        for (int i = tid; i < BN * BK; i += 256) {
            int n = i >> 5, k = i & 31;
            int gn = n0 + n, gk = k0 + k;
            Bs[n][k] = (gn < Nc && gk < K) ? Bm[(long)gk * b_rs + (long)gn * b_cs]
                                           : (bf16_t)0.0f;
        }
        // hint next tile into GL2 (global_prefetch_b8)
        if (k0 + BK < K && tid == 0) {
            __builtin_prefetch(&A[(long)m0 * a_rs + (long)(k0 + BK) * a_cs], 0, 1);
            __builtin_prefetch(&Bm[(long)(k0 + BK) * b_rs + (long)n0 * b_cs], 0, 1);
        }
        __syncthreads();

        // fragments per ISA layouts:
        //  A 16x32 bf16: lanes<16 hold K[0..7]+[16..23], lanes>=16 K[8..15]+[24..31]
        //  B 32x16 bf16: lanes<16 hold K[0..7]+[16..23], lanes>=16 K[8..15]+[24..31], lane%16 = N
        v16bf a_frag[2], b_frag[2];
        #pragma unroll
        for (int i = 0; i < 2; ++i) {
            const bf16_t* p = &As[wm + i * 16 + l16][half * 8];
            ((uint4*)&a_frag[i])[0] = *(const uint4*)p;
            ((uint4*)&a_frag[i])[1] = *(const uint4*)(p + 16);
        }
        #pragma unroll
        for (int j = 0; j < 2; ++j) {
            const bf16_t* p = &Bs[wn + j * 16 + l16][half * 8];
            ((uint4*)&b_frag[j])[0] = *(const uint4*)p;
            ((uint4*)&b_frag[j])[1] = *(const uint4*)(p + 16);
        }
        #pragma unroll
        for (int i = 0; i < 2; ++i)
            #pragma unroll
            for (int j = 0; j < 2; ++j) {
#if defined(__gfx1250__)
                acc[i][j] = __builtin_amdgcn_wmma_f32_16x16x32_bf16(
                    false, a_frag[i], false, b_frag[j],
                    (short)0, acc[i][j], false, false);
#else
                (void)a_frag; (void)b_frag;
#endif
            }
        __syncthreads();
    }

    // epilogue: C/D layout - VGPR r holds M=r (lanes 0-15) / M=r+8 (lanes 16-31)
    long cbase = (long)bz * c_bs;
    #pragma unroll
    for (int i = 0; i < 2; ++i)
        #pragma unroll
        for (int j = 0; j < 2; ++j)
            #pragma unroll
            for (int r = 0; r < 8; ++r) {
                int row = m0 + wm + i * 16 + half * 8 + r;
                int col = n0 + wn + j * 16 + l16;
                if (row < M && col < Nc) {
                    float v = acc[i][j][r];
                    if (bias) v += bias[row];
                    if (act == 1) v = geluf(v);
                    long idx = cbase + (long)row * c_rs + (long)col * c_cs;
                    if (Cb) Cb[idx] = (bf16_t)v; else C[idx] = v;
                }
            }
}

// ---------------------------------------------------------------------------
// Support kernels (bandwidth-bound glue)
// ---------------------------------------------------------------------------
__global__ void k_cvt(const float* __restrict__ in, bf16_t* __restrict__ out, long n) {
    long i = (long)blockIdx.x * 256 + threadIdx.x;
    if (i < n) out[i] = (bf16_t)in[i];
}

__global__ void k_copyf(float* __restrict__ out, const float* __restrict__ in, long n) {
    long i = (long)blockIdx.x * 256 + threadIdx.x;
    if (i < n) out[i] = in[i];
}

__global__ void k_addf(float* __restrict__ out, const float* __restrict__ a, long n) {
    long i = (long)blockIdx.x * 256 + threadIdx.x;
    if (i < n) out[i] += a[i];
}

// x[B,N,3] fp32 -> out[B,3,N] bf16
__global__ void k_transpose_x(const float* __restrict__ x, bf16_t* __restrict__ out) {
    long i = (long)blockIdx.x * 256 + threadIdx.x;
    if (i >= (long)BATCH * 3 * SEQ) return;
    long b = i / (3L * SEQ);
    long c = (i / SEQ) % 3;
    long n = i % SEQ;
    out[i] = (bf16_t)x[(b * SEQ + n) * 3 + c];
}

// BatchNorm training stats per channel over (batch, length)
__global__ void k_bn_stats(const float* __restrict__ x, float* __restrict__ mean,
                           float* __restrict__ var, int Nn, long bstride) {
    int c = blockIdx.x;
    __shared__ float s1[256], s2[256];
    float a = 0.f, q = 0.f;
    for (long i = threadIdx.x; i < (long)BATCH * Nn; i += 256) {
        long b = i / Nn, n = i % Nn;
        float v = x[b * bstride + (long)c * Nn + n];
        a += v; q += v * v;
    }
    s1[threadIdx.x] = a; s2[threadIdx.x] = q; __syncthreads();
    for (int s = 128; s > 0; s >>= 1) {
        if (threadIdx.x < s) { s1[threadIdx.x] += s1[threadIdx.x + s];
                               s2[threadIdx.x] += s2[threadIdx.x + s]; }
        __syncthreads();
    }
    if (threadIdx.x == 0) {
        float inv = 1.0f / ((float)BATCH * (float)Nn);
        float m = s1[0] * inv;
        mean[c] = m;
        var[c]  = s2[0] * inv - m * m;
    }
}

__global__ void k_bn_gelu_bf16(const float* __restrict__ x, const float* __restrict__ mean,
                               const float* __restrict__ var, const float* __restrict__ g,
                               const float* __restrict__ bt, bf16_t* __restrict__ out,
                               int Cc, int Nn, long out_bs, long total) {
    long i = (long)blockIdx.x * 256 + threadIdx.x;
    if (i >= total) return;
    long n = i % Nn;
    long t = i / Nn;
    int  c = (int)(t % Cc);
    long b = t / Cc;
    float y = g[c] * (x[i] - mean[c]) * rsqrtf(var[c] + 1e-5f) + bt[c];
    out[b * out_bs + (long)c * Nn + n] = (bf16_t)geluf(y);
}

// in-place row softmax over last dim (row length Nn), grid = B*Nn rows
__global__ void k_softmax(float* __restrict__ e, int Nn) {
    long row = blockIdx.x;
    float* r = e + row * (long)Nn;
    __shared__ float sh[256];
    float mx = -1e30f;
    for (int i = threadIdx.x; i < Nn; i += 256) mx = fmaxf(mx, r[i]);
    sh[threadIdx.x] = mx; __syncthreads();
    for (int s = 128; s > 0; s >>= 1) {
        if (threadIdx.x < s) sh[threadIdx.x] = fmaxf(sh[threadIdx.x], sh[threadIdx.x + s]);
        __syncthreads();
    }
    mx = sh[0]; __syncthreads();
    float sum = 0.f;
    for (int i = threadIdx.x; i < Nn; i += 256) { float v = __expf(r[i] - mx); r[i] = v; sum += v; }
    sh[threadIdx.x] = sum; __syncthreads();
    for (int s = 128; s > 0; s >>= 1) {
        if (threadIdx.x < s) sh[threadIdx.x] += sh[threadIdx.x + s];
        __syncthreads();
    }
    float inv = 1.0f / sh[0];
    for (int i = threadIdx.x; i < Nn; i += 256) r[i] *= inv;
}

// colsum[b][m] = sum_n e[b][n][m]
__global__ void k_colsum(const float* __restrict__ e, float* __restrict__ cs, int Nn) {
    long i = (long)blockIdx.x * 256 + threadIdx.x;
    if (i >= (long)BATCH * Nn) return;
    long b = i / Nn, m = i % Nn;
    const float* p = e + b * (long)Nn * Nn + m;
    float s = 0.f;
    for (int n = 0; n < Nn; ++n) s += p[(long)n * Nn];
    cs[i] = s;
}

__global__ void k_renorm(const float* __restrict__ e, const float* __restrict__ cs,
                         bf16_t* __restrict__ at, int Nn, long total) {
    long i = (long)blockIdx.x * 256 + threadIdx.x;
    if (i >= total) return;
    long m = i % Nn;
    long b = i / ((long)Nn * Nn);
    at[i] = (bf16_t)(e[i] / (1e-9f + cs[b * Nn + m]));
}

__global__ void k_sub_bf16(const bf16_t* __restrict__ x, long x_bs,
                           const float* __restrict__ xr, bf16_t* __restrict__ out,
                           long pc, long total) {
    long i = (long)blockIdx.x * 256 + threadIdx.x;
    if (i >= total) return;
    long b = i / pc, r = i % pc;
    out[i] = (bf16_t)((float)x[b * x_bs + r] - xr[i]);
}

__global__ void k_add_bf16(const bf16_t* __restrict__ x, long x_bs,
                           const bf16_t* __restrict__ r2, bf16_t* __restrict__ out,
                           long out_bs, long pc, long total) {
    long i = (long)blockIdx.x * 256 + threadIdx.x;
    if (i >= total) return;
    long b = i / pc, r = i % pc;
    out[b * out_bs + r] = (bf16_t)((float)x[b * x_bs + r] + (float)r2[i]);
}

__global__ void k_maxpool(const bf16_t* __restrict__ x, long x_bs, int Cc, int Nn,
                          float* __restrict__ out, long out_bs) {
    int i = blockIdx.x * 256 + threadIdx.x;
    if (i >= BATCH * Cc) return;
    int b = i / Cc, c = i % Cc;
    const bf16_t* p = x + (long)b * x_bs + (long)c * Nn;
    float m = -1e30f;
    for (int n = 0; n < Nn; ++n) m = fmaxf(m, (float)p[n]);
    out[(long)b * out_bs + c] = m;
}

__global__ void k_broadcast(const float* __restrict__ src, long src_bs,
                            bf16_t* __restrict__ dst, long dst_bs,
                            int Cc, int Nn, long total) {
    long i = (long)blockIdx.x * 256 + threadIdx.x;
    if (i >= total) return;
    long n = i % Nn;
    long t = i / Nn;
    long c = t % Cc;
    long b = t / Cc;
    dst[b * dst_bs + c * Nn + n] = (bf16_t)src[b * src_bs + c];
}

__global__ void k_copy_bf16(const bf16_t* __restrict__ in, long in_bs,
                            bf16_t* __restrict__ out, long out_bs, long pc, long total) {
    long i = (long)blockIdx.x * 256 + threadIdx.x;
    if (i >= total) return;
    long b = i / pc, r = i % pc;
    out[b * out_bs + r] = in[b * in_bs + r];
}

__global__ void k_ln(float* __restrict__ x, const float* __restrict__ g,
                     const float* __restrict__ b, int Dd) {
    int row = blockIdx.x;
    float* p = x + (long)row * Dd;
    __shared__ float s1[256], s2[256];
    float a = 0.f, q = 0.f;
    for (int i = threadIdx.x; i < Dd; i += 256) { float v = p[i]; a += v; q += v * v; }
    s1[threadIdx.x] = a; s2[threadIdx.x] = q; __syncthreads();
    for (int s = 128; s > 0; s >>= 1) {
        if (threadIdx.x < s) { s1[threadIdx.x] += s1[threadIdx.x + s];
                               s2[threadIdx.x] += s2[threadIdx.x + s]; }
        __syncthreads();
    }
    float m   = s1[0] / Dd;
    float var = s2[0] / Dd - m * m;
    float inv = rsqrtf(var + 1e-5f);
    for (int i = threadIdx.x; i < Dd; i += 256)
        p[i] = g[i] * (p[i] - m) * inv + b[i];
}

// ConvTranspose head epilogue: gelu(x + ps_b[o]) -> bf16 slice [B,128,128]
__global__ void k_bias_gelu_ps(const float* __restrict__ in, const float* __restrict__ bias,
                               bf16_t* __restrict__ out, long out_bs, long total) {
    long i = (long)blockIdx.x * 256 + threadIdx.x;
    if (i >= total) return;
    long b = i / 16384, m = i % 16384;
    int  o = (int)(m >> 7);
    out[b * out_bs + m] = (bf16_t)geluf(in[i] + bias[o]);
}

// ---------------------------------------------------------------------------
// Host-side orchestration
// ---------------------------------------------------------------------------
struct Scr {
    float  *G, *E, *CS, *MEAN, *VAR;
    bf16_t *QB, *VB, *ATT, *TB, *RB;
};

struct OAW {
    const bf16_t *Wq, *Wv, *Wt;
    const float  *vb, *tb, *g, *bt;
};

static void gemm(hipStream_t s, const bf16_t* A, const bf16_t* B, float* C, bf16_t* Cb,
                 const float* bias, int M, int Nc, int K, int z,
                 long a_rs, long a_cs, long a_bs,
                 long b_rs, long b_cs, long b_bs,
                 long c_rs, long c_cs, long c_bs, int act)
{
    dim3 g((unsigned)cdivl(Nc, BN), (unsigned)cdivl(M, BM), (unsigned)z);
    k_gemm_bf16<<<g, dim3(256), 0, s>>>(A, B, C, Cb, bias, M, Nc, K,
                                        a_rs, a_cs, a_bs, b_rs, b_cs, b_bs,
                                        c_rs, c_cs, c_bs, act);
}

static void conv_bn_gelu(hipStream_t s, const Scr& sc,
                         const bf16_t* x, long x_bs, int Cin, int Cout, int Nn,
                         const bf16_t* W, const float* bias,
                         const float* g, const float* bt,
                         bf16_t* out, long out_bs)
{
    gemm(s, W, x, sc.G, nullptr, bias, Cout, Nn, Cin, BATCH,
         Cin, 1, 0,   Nn, 1, x_bs,   Nn, 1, (long)Cout * Nn, 0);
    k_bn_stats<<<Cout, 256, 0, s>>>(sc.G, sc.MEAN, sc.VAR, Nn, (long)Cout * Nn);
    long tot = (long)BATCH * Cout * Nn;
    k_bn_gelu_bf16<<<(unsigned)cdivl(tot, 256), 256, 0, s>>>(
        sc.G, sc.MEAN, sc.VAR, g, bt, out, Cout, Nn, out_bs, tot);
}

static void run_oa(hipStream_t s, const Scr& sc, const OAW& w,
                   const bf16_t* x, long x_bs, bf16_t* out, long out_bs,
                   int C, int Nn)
{
    const int C4 = C / 4;
    // q = Wq x                       -> QB [B,C4,Nn] bf16
    gemm(s, w.Wq, x, nullptr, sc.QB, nullptr, C4, Nn, C, BATCH,
         C, 1, 0,   Nn, 1, x_bs,   Nn, 1, (long)C4 * Nn, 0);
    // v = Wv x + vb                  -> VB [B,C,Nn] bf16
    gemm(s, w.Wv, x, nullptr, sc.VB, w.vb, C, Nn, C, BATCH,
         C, 1, 0,   Nn, 1, x_bs,   Nn, 1, (long)C * Nn, 0);
    // energy = q^T q                 -> E [B,Nn,Nn] fp32 (transposed-A strides)
    gemm(s, sc.QB, sc.QB, sc.E, nullptr, nullptr, Nn, Nn, C4, BATCH,
         1, Nn, (long)C4 * Nn,   Nn, 1, (long)C4 * Nn,   Nn, 1, (long)Nn * Nn, 0);
    // softmax over keys, then L1 renorm over queries, emit bf16 attn
    k_softmax<<<BATCH * Nn, 256, 0, s>>>(sc.E, Nn);
    k_colsum<<<(unsigned)cdivl((long)BATCH * Nn, 256), 256, 0, s>>>(sc.E, sc.CS, Nn);
    long tot = (long)BATCH * Nn * Nn;
    k_renorm<<<(unsigned)cdivl(tot, 256), 256, 0, s>>>(sc.E, sc.CS, sc.ATT, Nn, tot);
    // x_r = v * attn                 -> G fp32
    gemm(s, sc.VB, sc.ATT, sc.G, nullptr, nullptr, C, Nn, Nn, BATCH,
         Nn, 1, (long)C * Nn,   Nn, 1, (long)Nn * Nn,   Nn, 1, (long)C * Nn, 0);
    // TB = x - x_r ; RB = gelu(bn(Wt TB + tb)) ; out = x + RB
    long tc = (long)BATCH * C * Nn;
    k_sub_bf16<<<(unsigned)cdivl(tc, 256), 256, 0, s>>>(x, x_bs, sc.G, sc.TB,
                                                        (long)C * Nn, tc);
    conv_bn_gelu(s, sc, sc.TB, (long)C * Nn, C, C, Nn, w.Wt, w.tb, w.g, w.bt,
                 sc.RB, (long)C * Nn);
    k_add_bf16<<<(unsigned)cdivl(tc, 256), 256, 0, s>>>(x, x_bs, sc.RB, out, out_bs,
                                                        (long)C * Nn, tc);
}

extern "C" void kernel_launch(void* const* d_in, const int* in_sizes, int n_in,
                              void* d_out, int out_size, void* d_ws, size_t ws_size,
                              hipStream_t stream)
{
    (void)in_sizes; (void)n_in; (void)out_size; (void)ws_size;
    auto in = [&](int i) { return (const float*)d_in[i]; };

    // ---- bump allocator over d_ws -----------------------------------------
    size_t off = 0;
    auto bump = [&](size_t bytes) -> void* {
        size_t a = (off + 255) & ~(size_t)255;
        off = a + bytes;
        return (void*)((char*)d_ws + a);
    };
    auto cvt = [&](const float* src, long n) -> bf16_t* {
        bf16_t* d = (bf16_t*)bump((size_t)n * sizeof(bf16_t));
        k_cvt<<<(unsigned)cdivl(n, 256), 256, 0, stream>>>(src, d, n);
        return d;
    };

    // ---- big scratch -------------------------------------------------------
    const long AB = (long)BATCH * 1024 * SEQ;            // max activation elems
    bf16_t* XT = (bf16_t*)bump((size_t)BATCH * 3 * SEQ * 2);
    bf16_t* A0 = (bf16_t*)bump((size_t)AB * 2);
    bf16_t* A1 = (bf16_t*)bump((size_t)AB * 2);
    bf16_t* A2 = (bf16_t*)bump((size_t)AB * 2);

    Scr sc;
    sc.G    = (float*)bump((size_t)AB * 4);
    sc.E    = (float*)bump((size_t)BATCH * SEQ * SEQ * 4);
    sc.ATT  = (bf16_t*)bump((size_t)BATCH * SEQ * SEQ * 2);
    sc.CS   = (float*)bump((size_t)BATCH * SEQ * 4);
    sc.MEAN = (float*)bump(1024 * 4);
    sc.VAR  = (float*)bump(1024 * 4);
    sc.QB   = (bf16_t*)bump((size_t)BATCH * 128 * SEQ * 2);
    sc.VB   = (bf16_t*)bump((size_t)AB * 2);
    sc.TB   = (bf16_t*)bump((size_t)AB * 2);
    sc.RB   = (bf16_t*)bump((size_t)AB * 2);

    // ---- weight conversion (input index map per setup_inputs order) -------
    // ext: idx 1..12 conv(W,b) x6 ; 13..24 bn(g,b) x6
    const int extCo[6] = {64, 128, 256, 512, 1024, 1024};
    const int extCi[6] = {3, 64, 128, 512, 512, 1024};
    bf16_t* extW[6];
    for (int i = 0; i < 6; ++i) extW[i] = cvt(in(1 + 2 * i), (long)extCo[i] * extCi[i]);

    auto loadOA = [&](int base, int C) -> OAW {
        OAW w;
        w.Wq = cvt(in(base),     (long)(C / 4) * C);
        w.Wv = cvt(in(base + 1), (long)C * C);
        w.vb = in(base + 2);
        w.Wt = cvt(in(base + 3), (long)C * C);
        w.tb = in(base + 4);
        w.g  = in(base + 5);
        w.bt = in(base + 6);
        return w;
    };

    // sal: 25 c1 | 27 bn1 | 29 oa64 | 36 c2 | 38 bn2 | 40 c3 | 42 bn3 |
    //      44 ca1 | 51 ca2 | 58 c4 | 60 bn4 | 62 c5 | 64 bn5 | 66 ca3 | 73 ca4 |
    //      80 c6 | 82 bn6 | 84 c7 | 86 bn7
    bf16_t* salC1 = cvt(in(25), 64L * 3);
    OAW oa64  = loadOA(29, 64);
    bf16_t* salC2 = cvt(in(36), 64L * 64);
    bf16_t* salC3 = cvt(in(40), 128L * 64);
    OAW ca1 = loadOA(44, 128);
    OAW ca2 = loadOA(51, 128);
    bf16_t* salC4 = cvt(in(58), 256L * 256);
    bf16_t* salC5 = cvt(in(62), 512L * 512);
    OAW ca3 = loadOA(66, 512);
    OAW ca4 = loadOA(73, 512);
    bf16_t* salC6 = cvt(in(80), 1024L * 1024);
    bf16_t* salC7 = cvt(in(84), 1024L * 1024);

    // ca: 88 proj(W,b) 90 ln13 92 ln12 94 Wqkv 95 bqkv 96 Wo 97 bo 98 l11 100 l12
    bf16_t* projW = cvt(in(88), 1024L * 1280);
    bf16_t* WvB   = cvt(in(94) + 2L * 1024 * 1024, 1024L * 1024);  // v-slice of Wqkv
    bf16_t* WoB   = cvt(in(96), 1024L * 1024);
    bf16_t* l11W  = cvt(in(98), 1024L * 1024);
    bf16_t* l12W  = cvt(in(100), 1024L * 1024);

    // head: 102 ps_W 103 ps_b 104 refuse(W,b) 106 oa1024 113 out1 115 out
    bf16_t* psW  = cvt(in(102), 1024L * 128 * 128);
    bf16_t* refW = cvt(in(104), 1024L * 1152);
    OAW oaH = loadOA(106, 1024);
    bf16_t* o1W  = cvt(in(113), 64L * 1536);
    bf16_t* ooW  = cvt(in(115), 3L * 64);

    // ---- small fp32/bf16 vectors ------------------------------------------
    float* EXTV = (float*)bump(BATCH * 1280 * 4);
    float *V1 = (float*)bump(4096 * 4), *V2 = (float*)bump(4096 * 4),
          *V3 = (float*)bump(4096 * 4), *V4 = (float*)bump(4096 * 4),
          *V5 = (float*)bump(4096 * 4);
    bf16_t *EXTVb = (bf16_t*)bump(5120 * 2), *SALVb = (bf16_t*)bump(5120 * 2);
    bf16_t *V1b = (bf16_t*)bump(4096 * 2), *V2b = (bf16_t*)bump(4096 * 2),
           *V3b = (bf16_t*)bump(4096 * 2), *Hb = (bf16_t*)bump(4096 * 2),
           *AVb = (bf16_t*)bump(4096 * 2);
    bf16_t* Cc1 = (bf16_t*)bump((size_t)BATCH * 1152 * 128 * 2);
    bf16_t* HF  = (bf16_t*)bump((size_t)BATCH * 1024 * 128 * 2);
    bf16_t* HO  = (bf16_t*)bump((size_t)BATCH * 1024 * 128 * 2);
    bf16_t* Cc2 = (bf16_t*)bump((size_t)BATCH * 1536 * 256 * 2);
    bf16_t* O1b = (bf16_t*)bump((size_t)BATCH * 64 * 256 * 2);

    float* OUT = (float*)d_out;           // [0,5120) sal | [5120,9216) adaptive | [9216,12288) coarse
    const long N = SEQ;

    // ---- input transpose: x[B,N,3] -> [B,3,N] bf16 ------------------------
    k_transpose_x<<<(unsigned)cdivl((long)BATCH * 3 * N, 256), 256, 0, stream>>>(in(0), XT);

    // =========================== _extensive ================================
    conv_bn_gelu(stream, sc, XT, 3 * N, 3, 64, (int)N, extW[0], in(2),  in(13), in(14), A0, 64 * N);
    conv_bn_gelu(stream, sc, A0, 64 * N, 64, 128, (int)N, extW[1], in(4),  in(15), in(16), A1, 128 * N);
    conv_bn_gelu(stream, sc, A1, 128 * N, 128, 256, (int)N, extW[2], in(6), in(17), in(18),
                 A2 + 256 * N, 512 * N);                        // into concat slot [256:512)
    k_maxpool<<<(unsigned)cdivl(BATCH * 256, 256), 256, 0, stream>>>(
        A2 + 256 * N, 512 * N, 256, (int)N, EXTV, 1280);        // ext low -> EXTV[:,0:256]
    k_broadcast<<<(unsigned)cdivl((long)BATCH * 256 * N, 256), 256, 0, stream>>>(
        EXTV, 1280, A2, 512 * N, 256, (int)N, (long)BATCH * 256 * N);
    conv_bn_gelu(stream, sc, A2, 512 * N, 512, 512, (int)N, extW[3], in(8),  in(19), in(20), A0, 512 * N);
    conv_bn_gelu(stream, sc, A0, 512 * N, 512, 1024, (int)N, extW[4], in(10), in(21), in(22), A1, 1024 * N);
    conv_bn_gelu(stream, sc, A1, 1024 * N, 1024, 1024, (int)N, extW[5], in(12), in(23), in(24), A2, 1024 * N);
    k_maxpool<<<(unsigned)cdivl(BATCH * 1024, 256), 256, 0, stream>>>(
        A2, 1024 * N, 1024, (int)N, EXTV + 256, 1280);          // ext fg -> EXTV[:,256:1280]

    // =========================== _salience =================================
    conv_bn_gelu(stream, sc, XT, 3 * N, 3, 64, (int)N, salC1, in(26), in(27), in(28), A0, 64 * N);
    run_oa(stream, sc, oa64, A0, 64 * N, A1, 64 * N, 64, (int)N);
    conv_bn_gelu(stream, sc, A1, 64 * N, 64, 64, (int)N, salC2, in(37), in(38), in(39), A0, 64 * N);
    conv_bn_gelu(stream, sc, A0, 64 * N, 64, 128, (int)N, salC3, in(41), in(42), in(43), A1, 128 * N);
    run_oa(stream, sc, ca1, A1, 128 * N, A2, 256 * N, 128, (int)N);                 // t0 -> ch[0:128)
    run_oa(stream, sc, ca2, A2, 256 * N, A2 + 128 * N, 256 * N, 128, (int)N);       // t1 -> ch[128:256)
    conv_bn_gelu(stream, sc, A2, 256 * N, 256, 256, (int)N, salC4, in(59), in(60), in(61),
                 A0 + 256 * N, 512 * N);                                            // f256 -> slot [256:512)
    k_maxpool<<<(unsigned)cdivl(BATCH * 256, 256), 256, 0, stream>>>(
        A0 + 256 * N, 512 * N, 256, (int)N, OUT, 1280);         // sal low -> d_out[:,0:256]
    k_broadcast<<<(unsigned)cdivl((long)BATCH * 256 * N, 256), 256, 0, stream>>>(
        OUT, 1280, A0, 512 * N, 256, (int)N, (long)BATCH * 256 * N);
    conv_bn_gelu(stream, sc, A0, 512 * N, 512, 512, (int)N, salC5, in(63), in(64), in(65), A1, 512 * N);
    run_oa(stream, sc, ca3, A1, 512 * N, A2, 1024 * N, 512, (int)N);                // t0 -> ch[0:512)
    run_oa(stream, sc, ca4, A2, 1024 * N, A2 + 512 * N, 1024 * N, 512, (int)N);     // t1 -> ch[512:1024)
    conv_bn_gelu(stream, sc, A2, 1024 * N, 1024, 1024, (int)N, salC6, in(81), in(82), in(83), A0, 1024 * N);
    conv_bn_gelu(stream, sc, A0, 1024 * N, 1024, 1024, (int)N, salC7, in(85), in(86), in(87), A1, 1024 * N);
    k_maxpool<<<(unsigned)cdivl(BATCH * 1024, 256), 256, 0, stream>>>(
        A1, 1024 * N, 1024, (int)N, OUT + 256, 1280);           // sal fg -> d_out[:,256:1280]

    // =========================== _cross ====================================
    k_cvt<<<(unsigned)cdivl(5120, 256), 256, 0, stream>>>(EXTV, EXTVb, 5120);
    k_cvt<<<(unsigned)cdivl(5120, 256), 256, 0, stream>>>(OUT,  SALVb, 5120);
    // s1 = LN13(proj f1), s2 = LN13(proj f2)   (columns = batch)
    gemm(stream, projW, EXTVb, V1, nullptr, in(89), 1024, BATCH, 1280, 1,
         1280, 1, 0,   1, 1280, 0,   1, 1024, 0, 0);
    k_ln<<<BATCH, 256, 0, stream>>>(V1, in(90), in(91), 1024);
    gemm(stream, projW, SALVb, V2, nullptr, in(89), 1024, BATCH, 1280, 1,
         1280, 1, 0,   1, 1280, 0,   1, 1024, 0, 0);
    k_ln<<<BATCH, 256, 0, stream>>>(V2, in(90), in(91), 1024);
    // src12 = Wo (Wv s2 + bv) + bo
    k_cvt<<<(unsigned)cdivl(4096, 256), 256, 0, stream>>>(V2, V2b, 4096);
    gemm(stream, WvB, V2b, V3, nullptr, in(95) + 2048, 1024, BATCH, 1024, 1,
         1024, 1, 0,   1, 1024, 0,   1, 1024, 0, 0);
    k_cvt<<<(unsigned)cdivl(4096, 256), 256, 0, stream>>>(V3, V3b, 4096);
    gemm(stream, WoB, V3b, V4, nullptr, in(97), 1024, BATCH, 1024, 1,
         1024, 1, 0,   1, 1024, 0,   1, 1024, 0, 0);
    // s1 = LN12(s1 + src12); s1 = s1 + l12(gelu(l11 s1))
    k_addf<<<(unsigned)cdivl(4096, 256), 256, 0, stream>>>(V1, V4, 4096);
    k_ln<<<BATCH, 256, 0, stream>>>(V1, in(92), in(93), 1024);
    k_cvt<<<(unsigned)cdivl(4096, 256), 256, 0, stream>>>(V1, V1b, 4096);
    gemm(stream, l11W, V1b, nullptr, Hb, in(99), 1024, BATCH, 1024, 1,
         1024, 1, 0,   1, 1024, 0,   1, 1024, 0, 1);
    gemm(stream, l12W, Hb, V5, nullptr, in(101), 1024, BATCH, 1024, 1,
         1024, 1, 0,   1, 1024, 0,   1, 1024, 0, 0);
    k_addf<<<(unsigned)cdivl(4096, 256), 256, 0, stream>>>(V1, V5, 4096);
    k_copyf<<<(unsigned)cdivl(4096, 256), 256, 0, stream>>>(OUT + 5120, V1, 4096);  // adaptive
    k_cvt<<<(unsigned)cdivl(4096, 256), 256, 0, stream>>>(V1, AVb, 4096);

    // =========================== head ======================================
    // ConvTranspose1d(1024,128,128) on length-1: out[b,m] = sum_c a[b,c] psW[c,m]
    gemm(stream, psW, AVb, sc.G, nullptr, nullptr, 16384, 1, 1024, BATCH,
         1, 16384, 0,   1, 1, 1024,   1, 1, 16384, 0);
    k_bias_gelu_ps<<<(unsigned)cdivl((long)BATCH * 16384, 256), 256, 0, stream>>>(
        sc.G, in(103), Cc1, 1152L * 128, (long)BATCH * 16384);              // ch[0:128)
    k_broadcast<<<(unsigned)cdivl((long)BATCH * 1024 * 128, 256), 256, 0, stream>>>(
        V1, 1024, Cc1 + 128L * 128, 1152L * 128, 1024, 128, (long)BATCH * 1024 * 128);
    // refuse conv + gelu
    gemm(stream, refW, Cc1, nullptr, HF, in(105), 1024, 128, 1152, BATCH,
         1152, 1, 0,   128, 1, 1152L * 128,   128, 1, 1024L * 128, 1);
    // offset attention on [B,1024,128]
    run_oa(stream, sc, oaH, HF, 1024L * 128, HO, 1024L * 128, 1024, 128);
    // x2 = reshape(.,[B,512,256]) into Cc2 ch[0:512); rep2 -> ch[512:1536)
    k_copy_bf16<<<(unsigned)cdivl((long)BATCH * 1024 * 128, 256), 256, 0, stream>>>(
        HO, 1024L * 128, Cc2, 1536L * 256, 1024L * 128, (long)BATCH * 1024 * 128);
    k_broadcast<<<(unsigned)cdivl((long)BATCH * 1024 * 256, 256), 256, 0, stream>>>(
        V1, 1024, Cc2 + 512L * 256, 1536L * 256, 1024, 256, (long)BATCH * 1024 * 256);
    // out1 (gelu) then out -> coarse fp32 straight into d_out
    gemm(stream, o1W, Cc2, nullptr, O1b, in(114), 64, 256, 1536, BATCH,
         1536, 1, 0,   256, 1, 1536L * 256,   256, 1, 64L * 256, 1);
    gemm(stream, ooW, O1b, OUT + 9216, nullptr, in(116), 3, 256, 64, BATCH,
         64, 1, 0,   256, 1, 64L * 256,   256, 1, 3L * 256, 0);
}
